// CausalSelfAttention_30210799960657
// MI455X (gfx1250) — matrix operand order repaired and checked by reading, not compiled, not run on hardware
//
#include <hip/hip_runtime.h>
#include <math.h>

// ---------------------------------------------------------------------------
// Causal self-attention, bf16 WMMA + async-LDS pipeline for gfx1250 (wave32).
// B=2, T=2048, C=1024, H=16, D=64.  Output = [y (B,T,C) | att (B,H,T,T)] fp32.
// ---------------------------------------------------------------------------

#define DEVINL __device__ __forceinline__

constexpr int kB = 2, kT = 2048, kC = 1024, kH = 16, kD = 64;
constexpr int kM = kB * kT;            // 4096 token rows
constexpr float kScale = 0.125f;       // 1/sqrt(64)

typedef __attribute__((ext_vector_type(16))) __bf16 v16bf;
typedef __attribute__((ext_vector_type(8)))  float  v8f;
typedef unsigned v4u __attribute__((ext_vector_type(4)));
typedef __attribute__((__vector_size__(16))) int v4i;   // async-LDS payload type

union ABFrag {
  unsigned       u32[8];
  unsigned short u16[16];
  v16bf          v;
};

// -------------------- CDNA5 async global->LDS copy path --------------------
#if defined(__has_builtin)
#  if __has_builtin(__builtin_amdgcn_global_load_async_to_lds_b128)
#    define ATHENA_HAVE_ASYNC_LDS 1
#  endif
#endif
#ifndef ATHENA_HAVE_ASYNC_LDS
#  define ATHENA_HAVE_ASYNC_LDS 0
#endif

DEVINL void async_copy16(const unsigned short* g, unsigned short* l) {
#if ATHENA_HAVE_ASYNC_LDS
  __builtin_amdgcn_global_load_async_to_lds_b128(
      (__attribute__((address_space(1))) v4i*)g,
      (__attribute__((address_space(3))) v4i*)l, 0, 0);
#else
  *(v4u*)l = *(const v4u*)g;                  // sync fallback (DScnt path)
#endif
}

DEVINL void async_wait_all() {
#if ATHENA_HAVE_ASYNC_LDS
#  if __has_builtin(__builtin_amdgcn_s_wait_asynccnt)
  __builtin_amdgcn_s_wait_asynccnt(0);
#  else
  asm volatile("s_wait_asynccnt 0" ::: "memory");
#  endif
#endif
}

// ----------------------------- small helpers -------------------------------
DEVINL unsigned short f32_to_bf16(float f) {
  unsigned u = __builtin_bit_cast(unsigned, f);
  u += 0x7FFFu + ((u >> 16) & 1u);     // round-to-nearest-even
  return (unsigned short)(u >> 16);
}

DEVINL v8f wmma_bf16(const ABFrag& a, const ABFrag& b, v8f c) {
  return __builtin_amdgcn_wmma_f32_16x16x32_bf16(false, a.v, false, b.v,
                                                 (short)0, c, false, false);
}

DEVINL v8f zero_v8f() {
  v8f c;
#pragma unroll
  for (int i = 0; i < 8; ++i) c[i] = 0.0f;
  return c;
}

// A-fragment (16x32 bf16) gather from a row-major [rows,K] bf16 matrix.
DEVINL void load_afrag(ABFrag& a, const unsigned short* row_base, int k0, int half) {
  const unsigned* pA = (const unsigned*)(row_base + k0 + half * 16);
#pragma unroll
  for (int t = 0; t < 4; ++t) { a.u32[t] = pA[t]; a.u32[4 + t] = pA[4 + t]; }
}

// B-fragment (32x16 bf16) from LDS tile with `ldk` shorts per row.
DEVINL void load_bfrag_lds(ABFrag& b, const unsigned short* lds_row0,
                           int n_local, int k_local, int ldk) {
  const unsigned* pB = (const unsigned*)(lds_row0 + (size_t)n_local * ldk + k_local);
#pragma unroll
  for (int t = 0; t < 8; ++t) b.u32[t] = pB[t];
}

// ---------------------------------------------------------------------------
// fp32 -> bf16 elementwise conversion
// ---------------------------------------------------------------------------
__global__ __launch_bounds__(256) void cvt_bf16_kernel(
    const float* __restrict__ src, unsigned short* __restrict__ dst, int n) {
  int i = blockIdx.x * 256 + threadIdx.x;
  if (i < n) dst[i] = f32_to_bf16(src[i]);
}

// ---------------------------------------------------------------------------
// D = A @ W^T + bias.  A: [M,K] bf16.  W: [N,K] bf16.
// Block = 8 waves, tile 256(M) x 64(N); wave tile 32x64 (2x4 accumulators).
// W panel [64n x 32k] (4 KB) staged to LDS with async DMA, double-buffered.
// mode 0: bf16 out [b,h,t,d] | mode 1: bf16 out [b,h,d,t] | mode 2: fp32 [M,N]
// ---------------------------------------------------------------------------
__global__ __launch_bounds__(256) void proj_gemm_kernel(
    const unsigned short* __restrict__ A, const unsigned short* __restrict__ W,
    const float* __restrict__ bias, void* __restrict__ out,
    int M, int N, int K, int mode) {
  __shared__ alignas(16) unsigned short ldsB[2][64 * 32];

  const int tid = threadIdx.x;
  const int wave = tid >> 5, lane = tid & 31;
  const int half = lane >> 4, lan16 = lane & 15;
  const int mBlk = blockIdx.x * 256;
  const int n0 = blockIdx.y * 64;

  // staging role: thread -> (row 0..63, 16B chunk 0..3) of the 64x32 panel
  const int srow = tid >> 2, schunk = tid & 3;
  const unsigned short* gB = W + (size_t)(n0 + srow) * K + schunk * 8;
  const int lofs = srow * 32 + schunk * 8;

  async_copy16(gB, &ldsB[0][lofs]);                       // prologue: k0 = 0

  const unsigned short* Arow0 = A + (size_t)(mBlk + wave * 32 + lan16) * K;
  const unsigned short* Arow1 = Arow0 + (size_t)16 * K;

  v8f acc[2][4];
#pragma unroll
  for (int mt = 0; mt < 2; ++mt)
#pragma unroll
    for (int nt = 0; nt < 4; ++nt) acc[mt][nt] = zero_v8f();

  for (int k0 = 0; k0 < K; k0 += 32) {
    const int cur = (k0 >> 5) & 1;
    async_wait_all();
    __syncthreads();                                      // buf[cur] ready
    if (k0 + 32 < K) async_copy16(gB + k0 + 32, &ldsB[cur ^ 1][lofs]);

    ABFrag a0, a1;
    load_afrag(a0, Arow0, k0, half);
    load_afrag(a1, Arow1, k0, half);
#pragma unroll
    for (int nt = 0; nt < 4; ++nt) {
      ABFrag b;
      load_bfrag_lds(b, ldsB[cur], nt * 16 + lan16, half * 16, 32);
      acc[0][nt] = wmma_bf16(a0, b, acc[0][nt]);
      acc[1][nt] = wmma_bf16(a1, b, acc[1][nt]);
    }
    __syncthreads();                                      // done reading buf[cur]
  }

#pragma unroll
  for (int mt = 0; mt < 2; ++mt) {
#pragma unroll
    for (int nt = 0; nt < 4; ++nt) {
#pragma unroll
      for (int r = 0; r < 8; ++r) {
        const int m = mBlk + wave * 32 + mt * 16 + r + 8 * half;
        const int n = n0 + nt * 16 + lan16;
        const float val = acc[mt][nt][r] + bias[n];
        if (mode == 2) {
          ((float*)out)[(size_t)m * N + n] = val;
        } else {
          const int bb = m / kT, t = m % kT, h = n >> 6, d = n & 63;
          const unsigned short bv = f32_to_bf16(val);
          if (mode == 0)
            ((unsigned short*)out)[((size_t)(bb * kH + h) * kT + t) * kD + d] = bv;
          else
            ((unsigned short*)out)[((size_t)(bb * kH + h) * kD + d) * kT + t] = bv;
        }
      }
    }
  }
}

// ---------------------------------------------------------------------------
// S = scale * Q @ K^T (causal), fp32 into att region of d_out.
// Block tile: 128(q) x 64(kk); wave tile 16 x 64 (A reused x4).
// K panel [64kk x 64d] (8 KB) staged once per block via async DMA.
// Upper-triangle blocks/waves skipped entirely.
// ---------------------------------------------------------------------------
__global__ __launch_bounds__(256) void qk_kernel(
    const unsigned short* __restrict__ Q,   // [BH, T, 64] bf16
    const unsigned short* __restrict__ Kh,  // [BH, T, 64] bf16
    float* __restrict__ att) {              // [BH, T, T]
  __shared__ alignas(16) unsigned short ldsK[64 * 64];

  const int tid = threadIdx.x;
  const int wave = tid >> 5, lane = tid & 31;
  const int half = lane >> 4, lan16 = lane & 15;
  const int kk0 = blockIdx.x * 64;
  const int qb0 = blockIdx.y * 128;
  if (kk0 > qb0 + 127) return;              // block fully masked (uniform)
  const int bh = blockIdx.z;
  const unsigned short* Qb = Q + (size_t)bh * kT * kD;
  const unsigned short* Kb = Kh + (size_t)bh * kT * kD;

  // stage K panel: 64 rows x 64 cols bf16 = 8 KB = 512 x 16B chunks
#pragma unroll
  for (int it = 0; it < 2; ++it) {
    const int idx = it * 256 + tid;
    const int row = idx >> 3, ch = idx & 7;
    async_copy16(Kb + (size_t)(kk0 + row) * kD + ch * 8, &ldsK[row * 64 + ch * 8]);
  }
  async_wait_all();
  __syncthreads();

  const int q0 = qb0 + wave * 16;
  if (kk0 > q0 + 15) return;                // wave fully masked (wave-uniform)

  v8f acc[4];
#pragma unroll
  for (int nt = 0; nt < 4; ++nt) acc[nt] = zero_v8f();

  const unsigned short* Arow = Qb + (size_t)(q0 + lan16) * kD;
#pragma unroll
  for (int k0 = 0; k0 < kD; k0 += 32) {
    ABFrag a;
    load_afrag(a, Arow, k0, half);
#pragma unroll
    for (int nt = 0; nt < 4; ++nt) {
      ABFrag b;
      load_bfrag_lds(b, ldsK, nt * 16 + lan16, k0 + half * 16, 64);
      acc[nt] = wmma_bf16(a, b, acc[nt]);
    }
  }

  float* Sb = att + (size_t)bh * kT * kT;
#pragma unroll
  for (int nt = 0; nt < 4; ++nt) {
#pragma unroll
    for (int r = 0; r < 8; ++r) {
      const int q = q0 + r + 8 * half;
      const int kk = kk0 + nt * 16 + lan16;
      float v = acc[nt][r] * kScale;
      if (kk > q) v = -__builtin_inff();
      Sb[(size_t)q * kT + kk] = v;
    }
  }
}

// ---------------------------------------------------------------------------
// Row softmax over j<=q; writes 0 for the masked tail (covers skipped tiles).
// ---------------------------------------------------------------------------
__global__ __launch_bounds__(256) void softmax_kernel(float* __restrict__ att) {
  __shared__ float red[256];
  const int tid = threadIdx.x;
  const int row = blockIdx.x;               // = bh*T + q
  const int q = row & (kT - 1);
  float* __restrict__ r = att + (size_t)row * kT;

  float m = -3.402823466e38f;
  for (int j = tid; j <= q; j += 256) m = fmaxf(m, r[j]);
  red[tid] = m; __syncthreads();
#pragma unroll
  for (int s = 128; s > 0; s >>= 1) {
    if (tid < s) red[tid] = fmaxf(red[tid], red[tid + s]);
    __syncthreads();
  }
  m = red[0]; __syncthreads();

  float sum = 0.0f;
  for (int j = tid; j <= q; j += 256) sum += __expf(r[j] - m);
  red[tid] = sum; __syncthreads();
#pragma unroll
  for (int s = 128; s > 0; s >>= 1) {
    if (tid < s) red[tid] += red[tid + s];
    __syncthreads();
  }
  const float inv = 1.0f / red[0];

  for (int j = tid; j < kT; j += 256)
    r[j] = (j <= q) ? __expf(r[j] - m) * inv : 0.0f;
}

// ---------------------------------------------------------------------------
// Y = P @ V.  P fp32 (bf16-converted in-register), V stored [BH,64,T] bf16.
// Block tile: 128(q) x 64(d); wave tile 16 x 64.  V panel [64d x 32kk] (4 KB)
// staged to LDS with async DMA, double-buffered; causal K bound per wave.
// ---------------------------------------------------------------------------
__global__ __launch_bounds__(256) void av_kernel(
    const float* __restrict__ att, const unsigned short* __restrict__ Vt,
    unsigned short* __restrict__ Yin) {
  __shared__ alignas(16) unsigned short ldsV[2][64 * 32];

  const int tid = threadIdx.x;
  const int wave = tid >> 5, lane = tid & 31;
  const int half = lane >> 4, lan16 = lane & 15;
  const int qb0 = blockIdx.x * 128;
  const int bh = blockIdx.y;
  const float* Pb = att + (size_t)bh * kT * kT;
  const unsigned short* Vb = Vt + (size_t)bh * kT * kD;   // [64 d][T kk]

  const int srow = tid >> 2, schunk = tid & 3;            // 64 rows x 4 chunks
  const unsigned short* gV = Vb + (size_t)srow * kT + schunk * 8;
  const int lofs = srow * 32 + schunk * 8;

  const int kend_blk = qb0 + 128;                         // block causal bound
  const int q0 = qb0 + wave * 16;
  const int my_kend = (q0 + 16 + 31) & ~31;               // wave causal bound

  async_copy16(gV, &ldsV[0][lofs]);                       // prologue: k0 = 0

  v8f acc[4];
#pragma unroll
  for (int nt = 0; nt < 4; ++nt) acc[nt] = zero_v8f();

  const float* pArow = Pb + (size_t)(q0 + lan16) * kT;
  for (int k0 = 0; k0 < kend_blk; k0 += 32) {
    const int cur = (k0 >> 5) & 1;
    async_wait_all();
    __syncthreads();
    if (k0 + 32 < kend_blk) async_copy16(gV + k0 + 32, &ldsV[cur ^ 1][lofs]);

    if (k0 < my_kend) {                                   // wave-uniform
      ABFrag a;
      const float* pa = pArow + k0 + half * 16;
#pragma unroll
      for (int j = 0; j < 8; ++j) {
        a.u16[j] = f32_to_bf16(pa[j]);
        a.u16[8 + j] = f32_to_bf16(pa[8 + j]);
      }
#pragma unroll
      for (int nt = 0; nt < 4; ++nt) {
        ABFrag b;
        load_bfrag_lds(b, ldsV[cur], nt * 16 + lan16, half * 16, 32);
        acc[nt] = wmma_bf16(a, b, acc[nt]);
      }
    }
    __syncthreads();
  }

  const int bb = bh / kH, h = bh % kH;
#pragma unroll
  for (int nt = 0; nt < 4; ++nt) {
#pragma unroll
    for (int r = 0; r < 8; ++r) {
      const int q = q0 + r + 8 * half;
      const int d = nt * 16 + lan16;
      Yin[(size_t)(bb * kT + q) * kC + h * kD + d] = f32_to_bf16(acc[nt][r]);
    }
  }
}

// ---------------------------------------------------------------------------
// Host-side launch
// ---------------------------------------------------------------------------
extern "C" void kernel_launch(void* const* d_in, const int* in_sizes, int n_in,
                              void* d_out, int out_size, void* d_ws, size_t ws_size,
                              hipStream_t stream) {
  (void)in_sizes; (void)n_in; (void)out_size; (void)ws_size;
  const float* x  = (const float*)d_in[0];
  const float* Wq = (const float*)d_in[1];
  const float* bq = (const float*)d_in[2];
  const float* Wk = (const float*)d_in[3];
  const float* bk = (const float*)d_in[4];
  const float* Wv = (const float*)d_in[5];
  const float* bv = (const float*)d_in[6];
  const float* Wo = (const float*)d_in[7];
  const float* bo = (const float*)d_in[8];

  float* y   = (float*)d_out;                          // [B,T,C]
  float* att = (float*)d_out + (size_t)kM * kC;        // [B,H,T,T]

  // Workspace carve-up (bf16 staging, 48 MB total).
  char* ws = (char*)d_ws;
  unsigned short* xb  = (unsigned short*)ws;            ws += (size_t)kM * kC * 2;
  unsigned short* Wqb = (unsigned short*)ws;            ws += (size_t)kC * kC * 2;
  unsigned short* Wkb = (unsigned short*)ws;            ws += (size_t)kC * kC * 2;
  unsigned short* Wvb = (unsigned short*)ws;            ws += (size_t)kC * kC * 2;
  unsigned short* Wob = (unsigned short*)ws;            ws += (size_t)kC * kC * 2;
  unsigned short* Qh  = (unsigned short*)ws;            ws += (size_t)kB * kH * kT * kD * 2;
  unsigned short* Kst = (unsigned short*)ws;            ws += (size_t)kB * kH * kT * kD * 2;
  unsigned short* Vt  = (unsigned short*)ws;            ws += (size_t)kB * kH * kT * kD * 2;
  unsigned short* Yin = (unsigned short*)ws;            ws += (size_t)kM * kC * 2;

  const int nx = kM * kC, nw = kC * kC;
  cvt_bf16_kernel<<<(nx + 255) / 256, 256, 0, stream>>>(x,  xb,  nx);
  cvt_bf16_kernel<<<(nw + 255) / 256, 256, 0, stream>>>(Wq, Wqb, nw);
  cvt_bf16_kernel<<<(nw + 255) / 256, 256, 0, stream>>>(Wk, Wkb, nw);
  cvt_bf16_kernel<<<(nw + 255) / 256, 256, 0, stream>>>(Wv, Wvb, nw);
  cvt_bf16_kernel<<<(nw + 255) / 256, 256, 0, stream>>>(Wo, Wob, nw);

  dim3 gProj(kM / 256, kC / 64);
  proj_gemm_kernel<<<gProj, 256, 0, stream>>>(xb, Wqb, bq, Qh,  kM, kC, kC, 0);
  proj_gemm_kernel<<<gProj, 256, 0, stream>>>(xb, Wkb, bk, Kst, kM, kC, kC, 0);
  proj_gemm_kernel<<<gProj, 256, 0, stream>>>(xb, Wvb, bv, Vt,  kM, kC, kC, 1);

  dim3 gQK(kT / 64, kT / 128, kB * kH);
  qk_kernel<<<gQK, 256, 0, stream>>>(Qh, Kst, att);

  softmax_kernel<<<kB * kH * kT, 256, 0, stream>>>(att);

  dim3 gAV(kT / 128, kB * kH);
  av_kernel<<<gAV, 256, 0, stream>>>(att, Vt, Yin);

  proj_gemm_kernel<<<gProj, 256, 0, stream>>>(Yin, Wob, bo, y, kM, kC, kC, 2);
}
